// PolygonGCN_87763361726773
// MI455X (gfx1250) — matrix operand order repaired and checked by reading, not compiled, not run on hardware
//
#include <hip/hip_runtime.h>

typedef _Float16 half_t;
typedef __attribute__((ext_vector_type(16))) _Float16 v16h;
typedef __attribute__((ext_vector_type(8)))  _Float16 v8h;
typedef __attribute__((ext_vector_type(8)))  float    v8f;

#define DEV __device__ __forceinline__

// ---------------------------------------------------------------------------
// LDS layout (dynamic shared, 318592 bytes total, <= 320KB WGP limit)
// ---------------------------------------------------------------------------
constexpr int OFF_H   = 0;                       // f32 [256][64]   activations (residual stream)
constexpr int OFF_HF  = OFF_H  + 256*64*4;       // f16 [256][72]   A-side copy of h
constexpr int OFF_Q   = OFF_HF + 256*72*2;       // f16 [256][72]   Q (later pool K)
constexpr int OFF_K   = OFF_Q  + 256*72*2;       // f16 [256][72]   K (later pool V)
constexpr int OFF_VT  = OFF_K  + 256*72*2;       // f16 [72][264]   V transposed (dim-major)
constexpr int OFF_P   = OFF_VT + 72*264*2;       // f16 8x[16][264] per-wave softmax(P) staging
constexpr int OFF_O   = OFF_P  + 8*16*264*2;     // f16 [256][72]   attention output
constexpr int SMEM_BYTES = OFF_O + 256*72*2;     // = 318592

// ---------------------------------------------------------------------------
// WMMA helpers (gfx1250 v_wmma_f32_16x16x32_f16)
// ---------------------------------------------------------------------------
DEV v8f wmma16(v16h a, v16h b, v8f c) {
  return __builtin_amdgcn_wmma_f32_16x16x32_f16(false, a, false, b, (short)0, c, false, false);
}

union UAB { v16h v; v8h h[2]; };

// A fragment, 16x32 f16 tile at (row0, k0) from row-major src with `stride` halves.
// ISA layout: lane<16 -> row m, K {k0..k0+7, k0+16..k0+23}; lane>=16 -> K {+8..+15, +24..+31}
DEV v16h loadA16(const half_t* src, int stride, int row0, int k0) {
  const int lane = threadIdx.x & 31;
  const int m = row0 + (lane & 15);
  const half_t* p = src + m * stride + k0 + ((lane < 16) ? 0 : 8);
  UAB u;
  u.h[0] = *(const v8h*)(p);
  u.h[1] = *(const v8h*)(p + 16);
  return u.v;
}

// B fragment, 32x16 f16 tile; bt is B-transposed row-major [N][K] with `stride` halves.
// ISA layout: lane<16 -> col n, K {k0..k0+7, k0+16..k0+23}; lane>=16 -> K {+8..+15, +24..+31}
DEV v16h loadBT16(const half_t* bt, int stride, int col0, int k0) {
  const int lane = threadIdx.x & 31;
  const int n = col0 + (lane & 15);
  const half_t* p = bt + n * stride + k0 + ((lane < 16) ? 0 : 8);
  UAB u;
  u.h[0] = *(const v8h*)(p);
  u.h[1] = *(const v8h*)(p + 16);
  return u.v;
}

// One-instruction block prefetch: fan 32 lanes across `bytes` starting at p.
// Lowers to global_prefetch_b8 (speculative, no counters touched).
DEV void prefetch_block(const void* p, int bytes) {
  const int lane = threadIdx.x & 31;
  const char* q = (const char*)p + (size_t)lane * (bytes >> 5);
  __builtin_prefetch(q, 0, 1);
}

// ---------------------------------------------------------------------------
// Parameter bundle
// ---------------------------------------------------------------------------
struct LayerW {
  const half_t *WqT, *WkT, *WvT, *WoT, *ff1T, *ff2T;   // f16 transposed weights in d_ws
  const float  *bq, *bk, *bv, *bo, *b1, *b2, *g1, *be1, *g2, *be2;
};
struct KArgs {
  const float *x, *perm;
  const float *initW, *initB;
  const float *convW[4], *convB[4];
  const float *gfcW, *gfcB;
  LayerW L[2];
  const float *normG, *normB;
  const half_t *poolWkT, *poolWvT;
  const float *poolBk, *poolBv, *poolWo, *poolBo;
  const float *qv;      // precomputed pool_q @ pool_Wq + pool_bq  (f32[64] in ws)
  float *out;
};

// ---------------------------------------------------------------------------
// Prep kernels: f32 [K][N] weight -> f16 transposed [N][K] in workspace
// ---------------------------------------------------------------------------
__global__ void k_transpose_f16(const float* __restrict__ W, half_t* __restrict__ out,
                                int K, int N) {
  int idx = blockIdx.x * 256 + threadIdx.x;
  if (idx >= K * N) return;
  int n = idx / K, k = idx % K;
  out[n * K + k] = (half_t)W[k * N + n];
}

__global__ void k_poolq(const float* __restrict__ pq, const float* __restrict__ Wq,
                        const float* __restrict__ bq, float* __restrict__ qv) {
  int j = threadIdx.x;
  if (j < 64) {
    float s = bq[j];
    for (int k = 0; k < 64; ++k) s += pq[k] * Wq[k * 64 + j];
    qv[j] = s;
  }
}

// ---------------------------------------------------------------------------
// Device building blocks (one workgroup = one polygon, 256 threads = 8 waves)
// ---------------------------------------------------------------------------

// dst = Asrc(256x64 f16, stride 72) @ W + bias ; W given as WT f16 [64][64]
DEV void gemm64_f16(const half_t* Asrc, const half_t* WT, const float* bias,
                    half_t* dst, int dstStride, bool transStore) {
  const int wv = threadIdx.x >> 5, lane = threadIdx.x & 31;
  const bool lo = lane < 16; const int l16 = lane & 15;
  for (int rt = 0; rt < 2; ++rt) {
    const int row0 = (2 * wv + rt) * 16;
#pragma unroll
    for (int ct = 0; ct < 4; ++ct) {
      v8f acc = {};
#pragma unroll
      for (int k0 = 0; k0 < 64; k0 += 32)
        acc = wmma16(loadA16(Asrc, 72, row0, k0), loadBT16(WT, 64, ct * 16, k0), acc);
      const int col = ct * 16 + l16;
      const float bb = bias[col];
      const int rb = row0 + (lo ? 0 : 8);
#pragma unroll
      for (int r = 0; r < 8; ++r) {
        float v = acc[r] + bb;
        if (!transStore) dst[(rb + r) * dstStride + col] = (half_t)v;
        else             dst[col * dstStride + (rb + r)] = (half_t)v;  // Vt layout
      }
    }
  }
}

// H += Asrc @ W + bias  (residual add, H f32 stride 64)
DEV void gemm64_resid(const half_t* Asrc, const half_t* WT, const float* bias, float* H) {
  const int wv = threadIdx.x >> 5, lane = threadIdx.x & 31;
  const bool lo = lane < 16; const int l16 = lane & 15;
  for (int rt = 0; rt < 2; ++rt) {
    const int row0 = (2 * wv + rt) * 16;
#pragma unroll
    for (int ct = 0; ct < 4; ++ct) {
      v8f acc = {};
#pragma unroll
      for (int k0 = 0; k0 < 64; k0 += 32)
        acc = wmma16(loadA16(Asrc, 72, row0, k0), loadBT16(WT, 64, ct * 16, k0), acc);
      const int col = ct * 16 + l16;
      const float bb = bias[col];
      const int rb = row0 + (lo ? 0 : 8);
#pragma unroll
      for (int r = 0; r < 8; ++r) H[(rb + r) * 64 + col] += acc[r] + bb;
    }
  }
}

// Multi-head self-attention core: per wave 2 row-tiles; per head hd=8 (K-padded frags)
DEV void attention(const half_t* Qb, const half_t* Kb, const half_t* Vt,
                   half_t* Pbase, half_t* Obuf) {
  const int wv = threadIdx.x >> 5, lane = threadIdx.x & 31;
  const bool lo = lane < 16; const int l16 = lane & 15;
  half_t* Pw = Pbase + wv * 16 * 264;
  const float scale = 0.3535533905932738f;  // 1/sqrt(8)
  const v8h z8 = {};
  const v8f zf = {};
  for (int rt = 0; rt < 2; ++rt) {
    const int row0 = (2 * wv + rt) * 16;
    for (int hh = 0; hh < 8; ++hh) {
      // A frag: Q rows, head dims 0..7 valid, K 8..31 zero-padded
      UAB ua; ua.h[0] = z8; ua.h[1] = z8;
      if (lo) ua.h[0] = *(const v8h*)(Qb + (row0 + l16) * 72 + 8 * hh);
      const v16h aq = ua.v;
      v8f sfr[16];
#pragma unroll
      for (int ct = 0; ct < 16; ++ct) sfr[ct] = zf;
#pragma unroll
      for (int ct = 0; ct < 16; ++ct) {
        UAB ub; ub.h[0] = z8; ub.h[1] = z8;     // B frag: K rows, K 8..31 zero
        if (lo) ub.h[0] = *(const v8h*)(Kb + (ct * 16 + l16) * 72 + 8 * hh);
        sfr[ct] = wmma16(aq, ub.v, sfr[ct]);
      }
      // softmax over 256 keys (row r lives in lanes 0-15, row r+8 in lanes 16-31)
      float rmax[8], rsum[8];
#pragma unroll
      for (int r = 0; r < 8; ++r) rmax[r] = -3.0e38f;
#pragma unroll
      for (int ct = 0; ct < 16; ++ct)
#pragma unroll
        for (int r = 0; r < 8; ++r) rmax[r] = fmaxf(rmax[r], sfr[ct][r]);
#pragma unroll
      for (int mk = 1; mk <= 8; mk <<= 1)
#pragma unroll
        for (int r = 0; r < 8; ++r) rmax[r] = fmaxf(rmax[r], __shfl_xor(rmax[r], mk, 32));
#pragma unroll
      for (int r = 0; r < 8; ++r) rsum[r] = 0.f;
#pragma unroll
      for (int ct = 0; ct < 16; ++ct)
#pragma unroll
        for (int r = 0; r < 8; ++r) {
          float e = __expf((sfr[ct][r] - rmax[r]) * scale);
          sfr[ct][r] = e;
          rsum[r] += e;
        }
#pragma unroll
      for (int mk = 1; mk <= 8; mk <<= 1)
#pragma unroll
        for (int r = 0; r < 8; ++r) rsum[r] += __shfl_xor(rsum[r], mk, 32);
      const int rb = lo ? 0 : 8;
#pragma unroll
      for (int ct = 0; ct < 16; ++ct)
#pragma unroll
        for (int r = 0; r < 8; ++r)
          Pw[(rb + r) * 264 + ct * 16 + l16] = (half_t)(sfr[ct][r] / rsum[r]);
      asm volatile("s_wait_dscnt 0" ::: "memory");  // P store -> A-frag load RAW
      // O_head = P @ V_head : B-frag rows Vt[8h+n][k]; cols 8..15 discarded
      v8f oacc = zf;
#pragma unroll
      for (int k0 = 0; k0 < 256; k0 += 32)
        oacc = wmma16(loadA16(Pw, 264, 0, k0), loadBT16(Vt, 264, 8 * hh, k0), oacc);
      if (l16 < 8) {
        const int orow = row0 + rb;
#pragma unroll
        for (int r = 0; r < 8; ++r)
          Obuf[(orow + r) * 72 + 8 * hh + l16] = (half_t)oacc[r];
      }
    }
  }
}

// FFN 64->2048->64, fused over 64-wide blocks; output accumulated in registers.
// Weight B-fragments stream straight from global (L2-resident after first WG);
// global_prefetch_b8 pulls the next jb block one iteration ahead.
DEV void ffn_block(const half_t* HF, half_t* Ubase, const LayerW& L, float* H) {
  const int wv = threadIdx.x >> 5, lane = threadIdx.x & 31;
  const bool lo = lane < 16; const int l16 = lane & 15;
  half_t* U = Ubase + wv * 32 * 72;  // wave-private [32][72] f16
  const v8f zf = {};
  v8f acc[2][4];
#pragma unroll
  for (int rt = 0; rt < 2; ++rt)
#pragma unroll
    for (int ct = 0; ct < 4; ++ct) acc[rt][ct] = zf;

  prefetch_block(L.ff1T, 8192);                       // jb = 0 block of W1^T

  for (int jb = 0; jb < 32; ++jb) {
    if (jb < 31) {
      prefetch_block(L.ff1T + (jb + 1) * 64 * 64, 8192);        // next W1^T block
      // next W2^T columns are strided (stride 2048 halves); touch one line per 4 rows
      prefetch_block(L.ff2T + (size_t)(l16 * 4) * 2048 + (jb + 1) * 64, 64);
    }
    // u = relu(h @ W1[:, jb*64 .. jb*64+64) + b1)
#pragma unroll
    for (int rt = 0; rt < 2; ++rt) {
      const int row0 = (2 * wv + rt) * 16;
#pragma unroll
      for (int ct = 0; ct < 4; ++ct) {
        v8f u = zf;
#pragma unroll
        for (int k0 = 0; k0 < 64; k0 += 32)
          u = wmma16(loadA16(HF, 72, row0, k0),
                     loadBT16(L.ff1T, 64, jb * 64 + ct * 16, k0), u);
        const int col = ct * 16 + l16;
        const float bb = L.b1[jb * 64 + col];
        const int rb = rt * 16 + (lo ? 0 : 8);
#pragma unroll
        for (int r = 0; r < 8; ++r)
          U[(rb + r) * 72 + col] = (half_t)fmaxf(u[r] + bb, 0.f);
      }
    }
    asm volatile("s_wait_dscnt 0" ::: "memory");  // U store -> A-frag load RAW
    // acc += u @ W2[jb*64 .. , :]
#pragma unroll
    for (int rt = 0; rt < 2; ++rt)
#pragma unroll
      for (int ct = 0; ct < 4; ++ct)
#pragma unroll
        for (int k0 = 0; k0 < 64; k0 += 32)
          acc[rt][ct] = wmma16(loadA16(U, 72, rt * 16, k0),
                               loadBT16(L.ff2T, 2048, ct * 16, jb * 64 + k0),
                               acc[rt][ct]);
  }
  // residual add + bias
#pragma unroll
  for (int rt = 0; rt < 2; ++rt)
#pragma unroll
    for (int ct = 0; ct < 4; ++ct) {
      const int col = ct * 16 + l16;
      const float bb = L.b2[col];
      const int rb = (2 * wv + rt) * 16 + (lo ? 0 : 8);
#pragma unroll
      for (int r = 0; r < 8; ++r) H[(rb + r) * 64 + col] += acc[rt][ct][r] + bb;
    }
}

DEV void ln_row(float* H, const float* g, const float* b) {
  const int r = threadIdx.x;
  float m = 0.f;
  for (int j = 0; j < 64; ++j) m += H[r * 64 + j];
  m *= (1.f / 64.f);
  float v = 0.f;
  for (int j = 0; j < 64; ++j) { float d = H[r * 64 + j] - m; v += d * d; }
  v *= (1.f / 64.f);
  const float inv = rsqrtf(v + 1e-5f);
  for (int j = 0; j < 64; ++j)
    H[r * 64 + j] = (H[r * 64 + j] - m) * inv * g[j] + b[j];
}

// ---------------------------------------------------------------------------
// Main fused kernel: one workgroup per polygon
// ---------------------------------------------------------------------------
__global__ __launch_bounds__(256) void poly_main(KArgs A) {
  extern __shared__ char smem[];
  float*  H   = (float*)(smem + OFF_H);
  half_t* HF  = (half_t*)(smem + OFF_HF);
  half_t* Qb  = (half_t*)(smem + OFF_Q);
  half_t* Kb  = (half_t*)(smem + OFF_K);
  half_t* Vt  = (half_t*)(smem + OFF_VT);
  half_t* Pb  = (half_t*)(smem + OFF_P);
  half_t* Ob  = (half_t*)(smem + OFF_O);
  const int tid = threadIdx.x, bidx = blockIdx.x;

  // ---- GCN stage (ring graph: deg==3 everywhere -> norm = 1/3) ----
  float* xcat = (float*)(smem + OFF_P);   // [256][64] f32 alias (P region)
  float* gt   = (float*)(smem + OFF_HF);  // [256][<=56] f32 alias (HF+Q region)
  {
    const float* xr = A.x + (bidx * 256 + tid) * 3;
    const float X0 = xr[0], X1 = xr[1], X2 = xr[2];
    for (int j = 0; j < 16; ++j) {
      float s = A.initB[j] + X0 * A.initW[j] + X1 * A.initW[16 + j] + X2 * A.initW[32 + j];
      xcat[tid * 64 + j] = fmaxf(s, 0.f);
    }
  }
  __syncthreads();
  for (int l = 0; l < 4; ++l) {
    const int inD = 16 * (l + 1), outD = (l < 3) ? 16 : 56;
    const float* W = A.convW[l];
    const float* bb = A.convB[l];
    for (int j = 0; j < outD; ++j) {
      float s = 0.f;
      for (int i = 0; i < inD; ++i) s += xcat[tid * 64 + i] * W[i * outD + j];
      gt[tid * outD + j] = s;
    }
    __syncthreads();
    const int pr = (tid + 255) & 255, nx = (tid + 1) & 255;
    for (int j = 0; j < outD; ++j) {
      float y = (gt[pr * outD + j] + gt[tid * outD + j] + gt[nx * outD + j]) * (1.f / 3.f) + bb[j];
      y = fmaxf(y, 0.f);
      if (l < 3) xcat[tid * 64 + inD + j] = y;
      else       H[tid * 64 + j] = y;
    }
    __syncthreads();
  }
  {
    const float pv = A.perm[bidx];
    for (int j = 0; j < 8; ++j) H[tid * 64 + 56 + j] = pv * A.gfcW[j] + A.gfcB[j];
  }
  // zero Vt pad rows 64..71 (read by head 7's discarded B-frag lanes)
  for (int idx = tid; idx < 8 * 264; idx += 256) Vt[64 * 264 + idx] = (half_t)0.f;
  __syncthreads();

  // ---- 2 transformer encoder layers (post-norm) ----
  for (int l = 0; l < 2; ++l) {
    const LayerW& L = A.L[l];
    prefetch_block(L.WqT, 8192);                 // Wq^T (8KB)
    prefetch_block(L.WvT, 8192);                 // Wv^T (8KB)
    for (int j = 0; j < 64; ++j) HF[tid * 72 + j] = (half_t)H[tid * 64 + j];
    __syncthreads();
    gemm64_f16(HF, L.WqT, L.bq, Qb, 72, false);
    gemm64_f16(HF, L.WkT, L.bk, Kb, 72, false);
    gemm64_f16(HF, L.WvT, L.bv, Vt, 264, true);    // V stored transposed [dim][seq]
    __syncthreads();
    attention(Qb, Kb, Vt, Pb, Ob);
    __syncthreads();
    gemm64_resid(Ob, L.WoT, L.bo, H);              // h += attn @ Wo + bo
    __syncthreads();
    ln_row(H, L.g1, L.be1);
    __syncthreads();
    for (int j = 0; j < 64; ++j) HF[tid * 72 + j] = (half_t)H[tid * 64 + j];
    __syncthreads();
    ffn_block(HF, Pb, L, H);                       // h += FFN(h) ; U aliases P region
    __syncthreads();
    ln_row(H, L.g2, L.be2);
    __syncthreads();
  }
  ln_row(H, A.normG, A.normB);
  __syncthreads();

  // ---- learned-query attention pooling ----
  for (int j = 0; j < 64; ++j) HF[tid * 72 + j] = (half_t)H[tid * 64 + j];
  __syncthreads();
  gemm64_f16(HF, A.poolWkT, A.poolBk, Qb, 72, false);   // Kp -> Qb
  gemm64_f16(HF, A.poolWvT, A.poolBv, Kb, 72, false);   // Vp -> Kb
  __syncthreads();
  float* sl = (float*)(smem + OFF_O);   // [8][256] logits/probs (alias O region)
  {
    const int n = tid;
    for (int h = 0; h < 8; ++h) {
      float s = 0.f;
      for (int d = 0; d < 8; ++d) s += A.qv[8 * h + d] * (float)Qb[n * 72 + 8 * h + d];
      sl[h * 256 + n] = s * 0.3535533905932738f;
    }
  }
  __syncthreads();
  {
    const int wv = tid >> 5, lane = tid & 31;   // wave w handles head w
    float* row = sl + wv * 256;
    float m = -3.0e38f;
    for (int n = lane; n < 256; n += 32) m = fmaxf(m, row[n]);
    for (int mk = 16; mk >= 1; mk >>= 1) m = fmaxf(m, __shfl_xor(m, mk, 32));
    float s = 0.f;
    for (int n = lane; n < 256; n += 32) { float e = __expf(row[n] - m); row[n] = e; s += e; }
    for (int mk = 16; mk >= 1; mk >>= 1) s += __shfl_xor(s, mk, 32);
    const float inv = 1.f / s;
    for (int n = lane; n < 256; n += 32) row[n] *= inv;
  }
  __syncthreads();
  float* attnv = (float*)(smem + OFF_P);
  if (tid < 64) {
    const int h = tid >> 3;
    float a = 0.f;
    for (int n = 0; n < 256; ++n) a += sl[h * 256 + n] * (float)Kb[n * 72 + tid];
    attnv[tid] = a;
  }
  __syncthreads();
  if (tid < 64) {
    float o = A.poolBo[tid];
    for (int d = 0; d < 64; ++d) o += attnv[d] * A.poolWo[d * 64 + tid];
    A.out[bidx * 64 + tid] = o;
  }
}

// ---------------------------------------------------------------------------
// Host launcher
// ---------------------------------------------------------------------------
extern "C" void kernel_launch(void* const* d_in, const int* in_sizes, int n_in,
                              void* d_out, int out_size, void* d_ws, size_t ws_size,
                              hipStream_t stream) {
  (void)in_sizes; (void)out_size; (void)ws_size;
  if (n_in < 60) return;
  auto F = [&](int i) { return (const float*)d_in[i]; };
  half_t* ws = (half_t*)d_ws;
  const size_t LSZ = 278528;  // halves per layer of transposed f16 weights

  struct TJ { int idx, K, N; size_t off; };
  const TJ jobs[14] = {
    {17, 64, 64, 0}, {18, 64, 64, 4096}, {19, 64, 64, 8192}, {20, 64, 64, 12288},
    {25, 64, 2048, 16384}, {27, 2048, 64, 147456},
    {33, 64, 64, LSZ + 0}, {34, 64, 64, LSZ + 4096}, {35, 64, 64, LSZ + 8192},
    {36, 64, 64, LSZ + 12288}, {41, 64, 2048, LSZ + 16384}, {43, 2048, 64, LSZ + 147456},
    {53, 64, 64, 2 * LSZ}, {54, 64, 64, 2 * LSZ + 4096},
  };
  for (int j = 0; j < 14; ++j) {
    const int total = jobs[j].K * jobs[j].N;
    k_transpose_f16<<<(total + 255) / 256, 256, 0, stream>>>(
        F(jobs[j].idx), ws + jobs[j].off, jobs[j].K, jobs[j].N);
  }
  float* qv = (float*)((char*)d_ws + (2 * LSZ + 8192) * 2);
  k_poolq<<<1, 64, 0, stream>>>(F(51), F(52), F(56), qv);

  KArgs A;
  A.x = F(0); A.perm = F(4);
  A.initW = F(5); A.initB = F(6);
  A.convW[0] = F(7);  A.convB[0] = F(8);
  A.convW[1] = F(9);  A.convB[1] = F(10);
  A.convW[2] = F(11); A.convB[2] = F(12);
  A.convW[3] = F(13); A.convB[3] = F(14);
  A.gfcW = F(15); A.gfcB = F(16);
  for (int l = 0; l < 2; ++l) {
    const int base = 17 + 16 * l;
    const size_t wo = (size_t)l * LSZ;
    A.L[l].WqT = ws + wo;          A.L[l].WkT = ws + wo + 4096;
    A.L[l].WvT = ws + wo + 8192;   A.L[l].WoT = ws + wo + 12288;
    A.L[l].ff1T = ws + wo + 16384; A.L[l].ff2T = ws + wo + 147456;
    A.L[l].bq = F(base + 4);  A.L[l].bk = F(base + 5);
    A.L[l].bv = F(base + 6);  A.L[l].bo = F(base + 7);
    A.L[l].b1 = F(base + 9);  A.L[l].b2 = F(base + 11);
    A.L[l].g1 = F(base + 12); A.L[l].be1 = F(base + 13);
    A.L[l].g2 = F(base + 14); A.L[l].be2 = F(base + 15);
  }
  A.normG = F(49); A.normB = F(50);
  A.poolWkT = ws + 2 * LSZ; A.poolWvT = ws + 2 * LSZ + 4096;
  A.poolBk = F(57); A.poolBv = F(58); A.poolWo = F(55); A.poolBo = F(59);
  A.qv = qv; A.out = (float*)d_out;

  poly_main<<<128, 256, SMEM_BYTES, stream>>>(A);
}
